// MultiheadAttention_84473416778109
// MI455X (gfx1250) — matrix operand order repaired and checked by reading, not compiled, not run on hardware
//
#include <hip/hip_runtime.h>

// ---------------------------------------------------------------------------
// MI455X (gfx1250, wave32) causal multi-head attention forward.
// Pipeline: f32->bf16 convert, Wt transposes, WMMA GEMM (x@Wqkv),
//           V transpose, flash attention (WMMA QK^T / PV, fp32 softmax),
//           WMMA GEMM (ctx@Wout + bias) -> fp32 out.
// All matmuls use v_wmma_f32_16x16x32_bf16. Loads are batch-issued ahead of
// the WMMAs that consume them so LOADcnt waits overlap with matrix work.
// ---------------------------------------------------------------------------

typedef __attribute__((ext_vector_type(4)))  __bf16 bf16x4;
typedef __attribute__((ext_vector_type(8)))  __bf16 bf16x8;
typedef __attribute__((ext_vector_type(16))) __bf16 bf16x16;
typedef __attribute__((ext_vector_type(4)))  float  f32x4;
typedef __attribute__((ext_vector_type(8)))  float  v8f;

#define DEV static __device__ __forceinline__

DEV bf16x16 cat16(bf16x8 a, bf16x8 b) {
  return __builtin_shufflevector(a, b, 0,1,2,3,4,5,6,7,8,9,10,11,12,13,14,15);
}
// Build a 16-element bf16 fragment from two 16-byte chunks.
DEV bf16x16 ld_frag(const __bf16* p0, const __bf16* p1) {
  return cat16(*(const bf16x8*)p0, *(const bf16x8*)p1);
}
DEV v8f wmma_bf16(bf16x16 a, bf16x16 b, v8f c) {
  // (neg_a, A, neg_b, B, c_mod, C, reuse_a, reuse_b)
  return __builtin_amdgcn_wmma_f32_16x16x32_bf16(false, a, false, b, (short)0, c,
                                                 false, false);
}

// ---------------------------------------------------------------------------
// Elementwise f32 -> bf16
// ---------------------------------------------------------------------------
__global__ void __launch_bounds__(256)
cvt_f32_to_bf16(const float* __restrict__ in, __bf16* __restrict__ out, int n) {
  int i = blockIdx.x * 256 + threadIdx.x;
  if (i < n) out[i] = (__bf16)in[i];
}

// ---------------------------------------------------------------------------
// Transpose + convert: in[R][C] f32 -> out[C][R] bf16 (R,C multiples of 32)
// ---------------------------------------------------------------------------
__global__ void __launch_bounds__(256)
transpose_cvt(const float* __restrict__ in, __bf16* __restrict__ out,
              int R, int C) {
  __shared__ float t[32][33];
  const int r0 = blockIdx.y * 32, c0 = blockIdx.x * 32;
  const int tx = threadIdx.x & 31, ty = threadIdx.x >> 5;
#pragma unroll
  for (int i = 0; i < 32; i += 8)
    t[ty + i][tx] = in[(size_t)(r0 + ty + i) * C + c0 + tx];
  __syncthreads();
#pragma unroll
  for (int i = 0; i < 32; i += 8)
    out[(size_t)(c0 + ty + i) * R + r0 + tx] = (__bf16)t[tx][ty + i];
}

// ---------------------------------------------------------------------------
// Extract V^T per (b,h): vt[(b*H+h)][d][n] = qkv[b*N+n][2D + h*128 + d]
// ---------------------------------------------------------------------------
__global__ void __launch_bounds__(256)
v_transpose(const __bf16* __restrict__ qkv, __bf16* __restrict__ vt) {
  __shared__ __bf16 t[32][40];
  const int bh = blockIdx.z;
  const int b = bh >> 4, h = bh & 15;
  const int n0 = blockIdx.x * 32, d0 = blockIdx.y * 32;
  const int tx = threadIdx.x & 31, ty = threadIdx.x >> 5;
  const __bf16* src =
      qkv + (size_t)(b * 2048 + n0) * 6144 + 2 * 2048 + h * 128 + d0;
#pragma unroll
  for (int i = 0; i < 32; i += 8)
    t[ty + i][tx] = src[(size_t)(ty + i) * 6144 + tx];
  __syncthreads();
  __bf16* dst = vt + ((size_t)bh * 128 + d0) * 2048 + n0;
#pragma unroll
  for (int i = 0; i < 32; i += 8)
    dst[(size_t)(ty + i) * 2048 + tx] = t[tx][ty + i];
}

// ---------------------------------------------------------------------------
// WMMA GEMM: C[M][N] = A[M][K] @ Bt[N][K]^T (+ bias), bf16 in / fp32 acc.
// Block: 256 threads (8 waves, 2x4 wave grid), tile 128x128x32.
// Two-stage pipeline: global->reg prefetch of tile k+1 overlaps the WMMAs of
// tile k (consumed from LDS). Epilogue stages C through per-wave LDS for
// coalesced b128/b64 stores.
// ---------------------------------------------------------------------------
template <typename OutT, bool BIAS>
__global__ void __launch_bounds__(256)
gemm_bf16_wmma(const __bf16* __restrict__ A, const __bf16* __restrict__ Bt,
               const float* __restrict__ bias, OutT* __restrict__ C,
               int M, int N, int K) {
  constexpr int BM = 128, BN = 128, BK = 32, LDT = BK + 8;  // 80B row stride
  __shared__ alignas(16) __bf16 As[BM * LDT];
  __shared__ alignas(16) __bf16 Bs[BN * LDT];
  __shared__ alignas(16) float cst[8 * 16 * 36];  // per-wave C staging
  const int tid = threadIdx.x;
  const int lane = tid & 31;
  const int wave = tid >> 5;
  const int wm = wave >> 2;  // 0..1 -> 64 rows each
  const int wn = wave & 3;   // 0..3 -> 32 cols each
  const int m0 = blockIdx.y * BM;
  const int n0 = blockIdx.x * BN;
  const int l15 = lane & 15;
  const bool lo = lane < 16;

  v8f acc[4][2] = {};

  const int c8 = (tid & 3) * 8;  // 16B chunk in K
  const int rr = tid >> 2;       // 0..63

  const __bf16* gA0 = &A[(size_t)(m0 + rr) * K + c8];
  const __bf16* gA1 = &A[(size_t)(m0 + rr + 64) * K + c8];
  const __bf16* gB0 = &Bt[(size_t)(n0 + rr) * K + c8];
  const __bf16* gB1 = &Bt[(size_t)(n0 + rr + 64) * K + c8];

  // prologue: tile 0 into registers
  bf16x8 ra0 = *(const bf16x8*)gA0;
  bf16x8 ra1 = *(const bf16x8*)gA1;
  bf16x8 rb0 = *(const bf16x8*)gB0;
  bf16x8 rb1 = *(const bf16x8*)gB1;

  for (int k0 = 0; k0 < K; k0 += BK) {
    // commit current tile to LDS
    *(bf16x8*)&As[(rr)*LDT + c8] = ra0;
    *(bf16x8*)&As[(rr + 64) * LDT + c8] = ra1;
    *(bf16x8*)&Bs[(rr)*LDT + c8] = rb0;
    *(bf16x8*)&Bs[(rr + 64) * LDT + c8] = rb1;
    __syncthreads();

    // issue next tile's global loads (uniform branch); they complete while
    // this tile's fragments + WMMAs execute
    if (k0 + BK < K) {
      ra0 = *(const bf16x8*)(gA0 + k0 + BK);
      ra1 = *(const bf16x8*)(gA1 + k0 + BK);
      rb0 = *(const bf16x8*)(gB0 + k0 + BK);
      rb1 = *(const bf16x8*)(gB1 + k0 + BK);
    }

    bf16x16 af[4], bfv[2];
#pragma unroll
    for (int mt = 0; mt < 4; ++mt) {
      // A 16x32 frag: lanes 0-15 rows, low half K 0..7|16..23, high 8..15|24..31
      const __bf16* r = &As[(wm * 64 + mt * 16 + l15) * LDT];
      const int ka = lo ? 0 : 8;
      af[mt] = ld_frag(r + ka, r + 16 + ka);
    }
#pragma unroll
    for (int nt = 0; nt < 2; ++nt) {
      // B 32x16 frag: lane = col, low half K 0..7|16..23, high 8..15|24..31
      const __bf16* r = &Bs[(wn * 32 + nt * 16 + l15) * LDT];
      const int kb = lo ? 0 : 8;
      bfv[nt] = ld_frag(r + kb, r + 16 + kb);
    }
#pragma unroll
    for (int mt = 0; mt < 4; ++mt)
#pragma unroll
      for (int nt = 0; nt < 2; ++nt)
        acc[mt][nt] = wmma_bf16(af[mt], bfv[nt], acc[mt][nt]);
    __syncthreads();
  }

  // ---- epilogue: stage per-wave 16x32 tiles in LDS, store coalesced ----
  float* st = &cst[wave * 16 * 36];
  const int srow = lane >> 3;        // 0..3
  const int scol = (lane & 7) * 4;   // 0,4,...,28
  float bv[2];
#pragma unroll
  for (int nt = 0; nt < 2; ++nt)
    bv[nt] = BIAS ? bias[n0 + wn * 32 + nt * 16 + l15] : 0.0f;

#pragma unroll
  for (int mt = 0; mt < 4; ++mt) {
#pragma unroll
    for (int nt = 0; nt < 2; ++nt) {
      const int col = nt * 16 + l15;
#pragma unroll
      for (int r = 0; r < 8; ++r)
        st[(r + (lo ? 0 : 8)) * 36 + col] = acc[mt][nt][r] + bv[nt];
    }
    // same-wave DS ops are in-order: reads below see the writes above
#pragma unroll
    for (int s = 0; s < 4; ++s) {
      const int row = srow + s * 4;  // 0..15
      f32x4 v4 = *(const f32x4*)&st[row * 36 + scol];
      OutT* gp = &C[(size_t)(m0 + wm * 64 + mt * 16 + row) * N + n0 +
                    wn * 32 + scol];
      if constexpr (sizeof(OutT) == 2) {
        bf16x4 o;
#pragma unroll
        for (int j = 0; j < 4; ++j) o[j] = (__bf16)v4[j];
        *(bf16x4*)gp = o;
      } else {
        *(f32x4*)gp = v4;
      }
    }
  }
}

// ---------------------------------------------------------------------------
// Flash attention (causal). One wave per 16-query tile, 8 waves/block.
// Computes S^T = K@Q^T so softmax's key reduction is per-lane + one xor-16
// shuffle, and ctx^T = V^T @ P accumulated in 8 fp32 WMMA fragments.
// K and V fragments for a key-step are batch-issued before any WMMA; V loads
// stay in flight behind the score WMMAs + softmax.
// ---------------------------------------------------------------------------
__global__ void __launch_bounds__(256)
flash_attn_wmma(const __bf16* __restrict__ qkv,  // [B*N, 3D]
                const __bf16* __restrict__ vt,   // [B*H][128][N]
                __bf16* __restrict__ ctx) {      // [B*N, D]
  constexpr int Nq = 2048, Dm = 2048, DHh = 128;
  constexpr int LD3 = 3 * Dm;
  const int lane = threadIdx.x & 31;
  const int wave = threadIdx.x >> 5;
  const int l15 = lane & 15;
  const bool lo = lane < 16;
  const int q0 = blockIdx.x * 128 + wave * 16;
  const int bh = blockIdx.y;
  const int b = bh >> 4;
  const int h = bh & 15;
  const float scale = 0.08838834764831845f;  // 1/sqrt(128)

  // Q^T B-fragments (lane = query col), 4 chunks of 32 dh, kept resident.
  const __bf16* Qrow = qkv + (size_t)(b * Nq + q0 + l15) * LD3 + h * DHh;
  bf16x16 qf[4];
#pragma unroll
  for (int c = 0; c < 4; ++c) {
    const __bf16* p = Qrow + c * 32 + (lo ? 0 : 8);
    qf[c] = ld_frag(p, p + 16);
  }

  v8f acc[8] = {};  // ctx^T fragments: 8 x (16 dh x 16 q)
  float mrun = -1e30f, lrun = 0.0f;

  const __bf16* Kb = qkv + (size_t)b * Nq * LD3 + Dm + h * DHh;
  const __bf16* Vtb = vt + (size_t)bh * DHh * Nq;
  const int qi = q0 + l15;

  for (int kt = 0; kt < q0 + 16; kt += 32) {
    // ---- batch-issue all K fragments (16 x b128) ----
    bf16x16 kf[8];
#pragma unroll
    for (int c = 0; c < 4; ++c) {
      const __bf16* p0 = Kb + (size_t)(kt + l15) * LD3 + c * 32 + (lo ? 0 : 8);
      kf[c] = ld_frag(p0, p0 + 16);
      const __bf16* p1 = p0 + (size_t)16 * LD3;
      kf[4 + c] = ld_frag(p1, p1 + 16);
    }
    // ---- batch-issue all V fragments (16 x b128); consumed after softmax ----
    bf16x16 vf[8];
#pragma unroll
    for (int f = 0; f < 8; ++f) {
      const __bf16* vp = Vtb + (size_t)(f * 16 + l15) * Nq + kt + (lo ? 0 : 16);
      vf[f] = ld_frag(vp, vp + 8);
    }
    // prefetch next key-step (uniform branch; global_prefetch_b8)
    if (kt + 32 < q0 + 16) {
      __builtin_prefetch(Kb + (size_t)(kt + 32 + l15) * LD3, 0, 1);
      __builtin_prefetch(Vtb + (size_t)l15 * Nq + kt + 32, 0, 1);
    }

    // ---- S^T tiles: keys kt..kt+15 (s0) and kt+16..kt+31 (s1) ----
    v8f s0 = {}, s1 = {};
#pragma unroll
    for (int c = 0; c < 4; ++c) s0 = wmma_bf16(kf[c], qf[c], s0);
#pragma unroll
    for (int c = 0; c < 4; ++c) s1 = wmma_bf16(kf[4 + c], qf[c], s1);

    // ---- scale + causal mask; row-max over 32 keys for this query ----
    float sv[16];
    float rmax = -1e30f;
#pragma unroll
    for (int r = 0; r < 8; ++r) {
      const int k0i = kt + r + (lo ? 0 : 8);
      const float x0 = (k0i <= qi) ? s0[r] * scale : -1e30f;
      const float x1 = (k0i + 16 <= qi) ? s1[r] * scale : -1e30f;
      sv[r] = x0;
      sv[8 + r] = x1;
      rmax = fmaxf(rmax, fmaxf(x0, x1));
    }
    rmax = fmaxf(rmax, __shfl_xor(rmax, 16));
    const float mnew = fmaxf(mrun, rmax);
    const float alpha = __expf(mrun - mnew);
    mrun = mnew;

    float pv[16], psum = 0.0f;
#pragma unroll
    for (int i = 0; i < 16; ++i) {
      pv[i] = __expf(sv[i] - mnew);
      psum += pv[i];
    }
    psum += __shfl_xor(psum, 16);
    lrun = lrun * alpha + psum;
#pragma unroll
    for (int f = 0; f < 8; ++f)
#pragma unroll
      for (int r = 0; r < 8; ++r) acc[f][r] *= alpha;

    // ---- repack P (C layout) into a 32x16 bf16 B-fragment ----
    // low lane needs keys 0..15: own tile0 + xor16(tile0)
    // high lane needs keys 16..31: xor16(tile1) + own tile1
    bf16x16 pf;
#pragma unroll
    for (int r = 0; r < 8; ++r) {
      const float s0x = __shfl_xor(pv[r], 16);
      const float s1x = __shfl_xor(pv[8 + r], 16);
      pf[r] = (__bf16)(lo ? pv[r] : s1x);
      pf[8 + r] = (__bf16)(lo ? s0x : pv[8 + r]);
    }

    // ---- ctx^T += V^T @ P : 8 dh tiles ----
#pragma unroll
    for (int f = 0; f < 8; ++f) acc[f] = wmma_bf16(vf[f], pf, acc[f]);
  }

  // ---- normalize and store ctx row-major: per lane 8 contiguous bf16 ----
  const float inv = 1.0f / lrun;
  __bf16* crow = ctx + (size_t)(b * Nq + q0 + l15) * Dm + h * DHh + (lo ? 0 : 8);
#pragma unroll
  for (int f = 0; f < 8; ++f) {
    bf16x8 o;
#pragma unroll
    for (int r = 0; r < 8; ++r) o[r] = (__bf16)(acc[f][r] * inv);
    *(bf16x8*)(crow + f * 16) = o;
  }
}

// ---------------------------------------------------------------------------
// Host launcher
// ---------------------------------------------------------------------------
extern "C" void kernel_launch(void* const* d_in, const int* in_sizes, int n_in,
                              void* d_out, int out_size, void* d_ws,
                              size_t ws_size, hipStream_t stream) {
  (void)in_sizes; (void)n_in; (void)out_size; (void)ws_size;
  constexpr int Bv = 4, Nv = 2048, Dv = 2048, Hv = 16;
  constexpr size_t TOK = (size_t)Bv * Nv;  // 8192 tokens

  const float* x = (const float*)d_in[0];
  const float* qkv_w = (const float*)d_in[1];
  const float* out_w = (const float*)d_in[2];
  const float* out_b = (const float*)d_in[3];
  float* out = (float*)d_out;

  char* ws = (char*)d_ws;
  __bf16* xb = (__bf16*)ws;    ws += TOK * Dv * 2;            // 32 MB
  __bf16* w1t = (__bf16*)ws;   ws += (size_t)3 * Dv * Dv * 2; // 24 MB
  __bf16* w2t = (__bf16*)ws;   ws += (size_t)Dv * Dv * 2;     //  8 MB
  __bf16* qkvb = (__bf16*)ws;  ws += TOK * 3 * Dv * 2;        // 96 MB
  __bf16* vtb = (__bf16*)ws;   ws += TOK * Dv * 2;            // 32 MB
  __bf16* ctxb = (__bf16*)ws;                                 // 32 MB

  {
    int n = (int)(TOK * Dv);
    cvt_f32_to_bf16<<<dim3((n + 255) / 256), 256, 0, stream>>>(x, xb, n);
  }
  transpose_cvt<<<dim3(3 * Dv / 32, Dv / 32), 256, 0, stream>>>(qkv_w, w1t, Dv,
                                                                3 * Dv);
  transpose_cvt<<<dim3(Dv / 32, Dv / 32), 256, 0, stream>>>(out_w, w2t, Dv, Dv);

  gemm_bf16_wmma<__bf16, false>
      <<<dim3(3 * Dv / 128, (int)(TOK / 128)), 256, 0, stream>>>(
          xb, w1t, nullptr, qkvb, (int)TOK, 3 * Dv, Dv);

  v_transpose<<<dim3(Nv / 32, 128 / 32, Bv * Hv), 256, 0, stream>>>(qkvb, vtb);

  flash_attn_wmma<<<dim3(Nv / 128, Bv * Hv), 256, 0, stream>>>(qkvb, vtb,
                                                               ctxb);

  gemm_bf16_wmma<float, true>
      <<<dim3(Dv / 128, (int)(TOK / 128)), 256, 0, stream>>>(
          ctxb, w2t, out_b, out, (int)TOK, Dv, Dv);
}
